// OnlineDFlashModel_68762426409727
// MI455X (gfx1250) — matrix operand order repaired and checked by reading, not compiled, not run on hardware
//
#include <hip/hip_runtime.h>
#include <cstdint>

// ---------------- problem constants (from reference) ----------------
constexpr int Bc   = 2;
constexpr int Sc   = 2048;
constexpr int BSc  = 16;
constexpr int Nc   = 64;
constexpr int Vc   = 32000;
constexpr int Dc   = 1024;
constexpr int Hc   = 8;
constexpr int DHc  = 128;          // D / H
constexpr int QL   = Nc * BSc;     // 1024
constexpr int KVL  = Sc + QL;      // 3072
constexpr int MASKID = 3;
constexpr float GAMMAc = 2.0f;
constexpr float SCALEc = 0.08838834764831845f;   // 1/sqrt(128)
constexpr int VSPLIT   = 20;
constexpr int VTILES64 = Vc / 64;  // 500 (64-wide column tiles)
constexpr int ROWS     = Bc * QL;  // 2048 logit rows

// ---------------- WMMA types ----------------
typedef __attribute__((ext_vector_type(16))) __bf16 v16bf;
typedef __attribute__((ext_vector_type(8)))  float  v8f;

union BF16x16 { v16bf v; uint4 u[2]; unsigned short s[16]; };

__device__ __forceinline__ unsigned short f2bf(float f) {
  unsigned u = __float_as_uint(f);
  unsigned r = u + 0x7FFFu + ((u >> 16) & 1u);   // round-to-nearest-even
  return (unsigned short)(r >> 16);
}
__device__ __forceinline__ float bf2f(unsigned short h) {
  return __uint_as_float(((unsigned)h) << 16);
}

// =====================================================================
// 0) zero the two loss accumulators
// =====================================================================
__global__ void zero_accum_kernel(float* acc) {
  if (threadIdx.x < 2) acc[threadIdx.x] = 0.f;
}

// =====================================================================
// 1) build kv_in = concat(ctx_emb, noise_emb) as bf16 [B, KVL, D]
// =====================================================================
__global__ __launch_bounds__(256) void prep_kv_kernel(
    const int* __restrict__ input_ids, const int* __restrict__ anchors,
    const unsigned char* __restrict__ keep_mask, const float* __restrict__ embed,
    unsigned short* __restrict__ kv_bf16)
{
  int pos = blockIdx.x;                 // 0 .. B*KVL-1
  int b = pos / KVL, p = pos % KVL;
  int tok;
  if (p < Sc) {
    tok = input_ids[b * Sc + p];
  } else {
    int q = p - Sc;
    if ((q & (BSc - 1)) != 0) tok = MASKID;
    else {
      int n = q / BSc;
      int a = anchors[b * Nc + n];
      a = min(max(a, 0), Sc - 1);
      tok = keep_mask[b * Nc + n] ? input_ids[b * Sc + a] : MASKID;
    }
  }
  const float4* e = (const float4*)(embed + (size_t)tok * Dc);
  ushort4* o = (ushort4*)(kv_bf16 + (size_t)pos * Dc);
  float4 x = e[threadIdx.x];
  ushort4 y; y.x = f2bf(x.x); y.y = f2bf(x.y); y.z = f2bf(x.z); y.w = f2bf(x.w);
  o[threadIdx.x] = y;
}

// =====================================================================
// 2) transpose-convert weights: src f32 [R][C] -> dst bf16 [C][R]
// =====================================================================
__global__ void tconv_kernel(const float* __restrict__ src,
                             unsigned short* __restrict__ dst, int R, int C)
{
  __shared__ float tile[32][33];
  int c0 = blockIdx.x * 32, r0 = blockIdx.y * 32;
  int tx = threadIdx.x, ty = threadIdx.y;   // block (32, 8)
  for (int i = 0; i < 4; ++i) {
    int r = r0 + ty + i * 8, c = c0 + tx;
    tile[ty + i * 8][tx] = (r < R && c < C) ? src[(size_t)r * C + c] : 0.f;
  }
  __syncthreads();
  for (int i = 0; i < 4; ++i) {
    int c = c0 + ty + i * 8, r = r0 + tx;
    if (c < C && r < R) dst[(size_t)c * R + r] = f2bf(tile[tx][ty + i * 8]);
  }
}

// =====================================================================
// 3) register-blocked bf16 WMMA GEMM:  C[M,N] = A[M,K] * Bt[N,K]^T
//    one wave per 64x64 C block: 4x4 grid of 16x16 WMMA tiles
//    (16 v8f accumulators; 4 A-frags + 4 B-frags reused 4x each per K step)
//    optional transposed bf16 store (for V^T)
// =====================================================================
__global__ __launch_bounds__(32) void gemm_bf16_kernel(
    const unsigned short* __restrict__ A, const unsigned short* __restrict__ Bt,
    unsigned short* __restrict__ C,
    int M, int Nn, int K, long sA, long sC, int transC, int ldT)
{
  int tn = blockIdx.x * 64;
  int tm = blockIdx.y * 64;
  int b  = blockIdx.z;
  const unsigned short* Ab = A + (size_t)b * sA;
  unsigned short* Cp = C + (size_t)b * sC;
  int lane = threadIdx.x, half = lane >> 4, l16 = lane & 15;

  v8f acc[16];
  for (int i = 0; i < 16; ++i) acc[i] = {};

  const unsigned short* arow[4];
  const unsigned short* brow[4];
  for (int i = 0; i < 4; ++i) {
    arow[i] = Ab + (size_t)(tm + i * 16 + l16) * K;
    brow[i] = Bt + (size_t)(tn + i * 16 + l16) * K;
  }
  for (int k0 = 0; k0 < K; k0 += 32) {
    BF16x16 af[4], bf[4];
    int ca = k0 + (half ? 8 : 0);
    int cb = k0 + (half ? 8 : 0);
    for (int i = 0; i < 4; ++i) {
      af[i].u[0] = *(const uint4*)(arow[i] + ca);
      af[i].u[1] = *(const uint4*)(arow[i] + ca + 16);
      bf[i].u[0] = *(const uint4*)(brow[i] + cb);
      bf[i].u[1] = *(const uint4*)(brow[i] + cb + 16);
    }
    for (int mi = 0; mi < 4; ++mi)
      for (int ni = 0; ni < 4; ++ni)
        acc[mi * 4 + ni] = __builtin_amdgcn_wmma_f32_16x16x32_bf16(
            false, af[mi].v, false, bf[ni].v, (short)0, acc[mi * 4 + ni],
            false, false);
  }
  if (!transC) {
    for (int mi = 0; mi < 4; ++mi)
      for (int ni = 0; ni < 4; ++ni)
        for (int j = 0; j < 8; ++j) {
          int row = (half ? 8 : 0) + j;
          Cp[(size_t)(tm + mi * 16 + row) * Nn + tn + ni * 16 + l16] =
              f2bf(acc[mi * 4 + ni][j]);
        }
  } else {  // C^T store: per tile, 8 consecutive m per lane -> one b128 store
    for (int mi = 0; mi < 4; ++mi)
      for (int ni = 0; ni < 4; ++ni) {
        union { unsigned short s[8]; uint4 u; } t;
        for (int j = 0; j < 8; ++j) t.s[j] = f2bf(acc[mi * 4 + ni][j]);
        unsigned short* cp = Cp + (size_t)(tn + ni * 16 + l16) * ldT
                             + tm + mi * 16 + (half ? 8 : 0);
        *(uint4*)cp = t.u;
      }
  }
}

// =====================================================================
// 4) flash attention: one wave per (b, h, 16-row q block)
//    q [B,QL,D] bf16, k [B,KVL,D] bf16, vT [B,D,KVL] bf16 -> attn [B,QL,D] bf16
// =====================================================================
__global__ __launch_bounds__(32) void attn_kernel(
    const unsigned short* __restrict__ qb, const unsigned short* __restrict__ kb,
    const unsigned short* __restrict__ vT, const int* __restrict__ anchors,
    unsigned short* __restrict__ attn_out)
{
  int nblk = blockIdx.x;   // q block == anchor index (BS == 16)
  int h    = blockIdx.y;
  int b    = blockIdx.z;
  int lane = threadIdx.x, half = lane >> 4, l16 = lane & 15;

  __shared__ __align__(16) float          s_sc[16 * 32];
  __shared__ __align__(16) unsigned short s_p[16 * 32];
  __shared__ float s_m[16], s_l[16], s_f[16];

  int anch = anchors[b * Nc + nblk];
  if (lane < 16) { s_m[lane] = -3.0e38f; s_l[lane] = 0.f; }
  __syncthreads();

  v8f acc[8];
  for (int t = 0; t < 8; ++t) acc[t] = {};

  const unsigned short* qbase = qb + ((size_t)(b * QL + nblk * 16)) * Dc + h * DHc;
  const unsigned short* kbase = kb + ((size_t)b * KVL) * Dc + h * DHc;
  const unsigned short* vbase = vT + ((size_t)(b * Dc + h * DHc)) * KVL;

  int nctx = (anch + 31) / 32;
  for (int s = 0; s <= nctx; ++s) {
    bool draft = (s == nctx);
    int kv0 = draft ? (Sc + nblk * 16 - 16) : s * 32;

    // ---- scores: two 16x16 tiles over K = DH = 128 ----
    v8f sc0 = {}, sc1 = {};
    for (int ks = 0; ks < 4; ++ks) {
      int k0 = ks * 32;
      BF16x16 af;
      {
        int ca = k0 + (half ? 8 : 0);
        const unsigned short* ap = qbase + (size_t)l16 * Dc + ca;
        af.u[0] = *(const uint4*)ap;
        af.u[1] = *(const uint4*)(ap + 16);
      }
      for (int t2 = 0; t2 < 2; ++t2) {
        BF16x16 bf;
        int kvcol = kv0 + t2 * 16 + l16;
        const unsigned short* bp = kbase + (size_t)kvcol * Dc + k0 + (half ? 8 : 0);
        bf.u[0] = *(const uint4*)bp;
        bf.u[1] = *(const uint4*)(bp + 16);
        if (t2 == 0)
          sc0 = __builtin_amdgcn_wmma_f32_16x16x32_bf16(false, af.v, false, bf.v,
                                                        (short)0, sc0, false, false);
        else
          sc1 = __builtin_amdgcn_wmma_f32_16x16x32_bf16(false, af.v, false, bf.v,
                                                        (short)0, sc1, false, false);
      }
    }
    for (int j = 0; j < 8; ++j) {
      int row = (half ? 8 : 0) + j;
      s_sc[row * 32 + l16]      = sc0[j];
      s_sc[row * 32 + 16 + l16] = sc1[j];
    }
    __syncthreads();

    // ---- per-row online softmax (lanes 0..15, one row each) ----
    if (lane < 16) {
      float m_old = s_m[lane], mx = m_old;
      for (int j = 0; j < 32; ++j) {
        bool ok = draft ? (j >= 16) : ((kv0 + j) < anch);
        float v = ok ? s_sc[lane * 32 + j] * SCALEc : -3.0e38f;
        mx = fmaxf(mx, v);
      }
      float f = __expf(m_old - mx);
      float sum = 0.f;
      for (int j = 0; j < 32; ++j) {
        bool ok = draft ? (j >= 16) : ((kv0 + j) < anch);
        float p = ok ? __expf(s_sc[lane * 32 + j] * SCALEc - mx) : 0.f;
        sum += p;
        s_p[lane * 32 + j] = f2bf(p);
      }
      s_l[lane] = s_l[lane] * f + sum;
      s_m[lane] = mx;
      s_f[lane] = f;
    }
    __syncthreads();

    // ---- rescale accumulators, then P (16x32) x V (32x128) ----
    float fr[8];
    for (int j = 0; j < 8; ++j) fr[j] = s_f[(half ? 8 : 0) + j];
    for (int t = 0; t < 8; ++t)
      for (int j = 0; j < 8; ++j) acc[t][j] *= fr[j];

    BF16x16 pf;
    {
      const unsigned short* pp = &s_p[l16 * 32 + (half ? 8 : 0)];
      pf.u[0] = *(const uint4*)pp;
      pf.u[1] = *(const uint4*)(pp + 16);
    }
    for (int t = 0; t < 8; ++t) {
      BF16x16 bf;
      const unsigned short* bp =
          vbase + (size_t)(t * 16 + l16) * KVL + kv0 + (half ? 8 : 0);
      bf.u[0] = *(const uint4*)bp;
      bf.u[1] = *(const uint4*)(bp + 16);
      acc[t] = __builtin_amdgcn_wmma_f32_16x16x32_bf16(false, pf.v, false, bf.v,
                                                       (short)0, acc[t], false, false);
    }
  }

  // ---- finalize: divide by l, store bf16 ----
  float linv[8];
  for (int j = 0; j < 8; ++j)
    linv[j] = 1.f / fmaxf(s_l[(half ? 8 : 0) + j], 1e-30f);
  for (int t = 0; t < 8; ++t)
    for (int j = 0; j < 8; ++j) {
      int row = (half ? 8 : 0) + j;
      int q = nblk * 16 + row;
      attn_out[((size_t)(b * QL + q)) * Dc + h * DHc + t * 16 + l16] =
          f2bf(acc[t][j] * linv[j]);
    }
}

// =====================================================================
// 5) LM head fused with streaming logsumexp (partials over V slices)
//    hid [2048,1024] bf16 * Wt_lm[V,1024]^T ; no logits materialized.
//    block = 4 waves; wave w owns rows rb*64 + w*16 .. +15
//    each pass computes a 16x64 logit block (1 shared A-frag, 4 B-frags,
//    4 WMMAs per K step), then a 64-wide online logsumexp update.
// =====================================================================
__global__ __launch_bounds__(128) void lm_lse_kernel(
    const unsigned short* __restrict__ Hb, const unsigned short* __restrict__ Wt,
    float* __restrict__ part_m, float* __restrict__ part_l)
{
  int rb = blockIdx.x;                 // 0..31
  int vs = blockIdx.y;                 // 0..VSPLIT-1
  int wave = threadIdx.x >> 5;
  int lane = threadIdx.x & 31, half = lane >> 4, l16 = lane & 15;
  int r0 = rb * 64 + wave * 16;

  __shared__ __align__(16) float s_sc[4][16 * 64];

  const unsigned short* arow = Hb + (size_t)(r0 + l16) * Dc;
  float m_i = -3.0e38f, l_i = 0.f;     // per-row state held in lanes 0..15

  int ct0 = vs * (VTILES64 / VSPLIT), ct1 = ct0 + (VTILES64 / VSPLIT);
  for (int ct = ct0; ct < ct1; ++ct) {
    int v0 = ct * 64;
    v8f acc[4];
    for (int i = 0; i < 4; ++i) acc[i] = {};
    const unsigned short* brow[4];
    for (int i = 0; i < 4; ++i)
      brow[i] = Wt + (size_t)(v0 + i * 16 + l16) * Dc;
    for (int k0 = 0; k0 < Dc; k0 += 32) {
      BF16x16 af, bf[4];
      int ca = k0 + (half ? 8 : 0);
      af.u[0] = *(const uint4*)(arow + ca);
      af.u[1] = *(const uint4*)(arow + ca + 16);
      int cb = k0 + (half ? 8 : 0);
      for (int i = 0; i < 4; ++i) {
        __builtin_prefetch(brow[i] + cb + 256, 0, 1);   // global_prefetch_b8
        bf[i].u[0] = *(const uint4*)(brow[i] + cb);
        bf[i].u[1] = *(const uint4*)(brow[i] + cb + 16);
      }
      for (int i = 0; i < 4; ++i)
        acc[i] = __builtin_amdgcn_wmma_f32_16x16x32_bf16(
            false, af.v, false, bf[i].v, (short)0, acc[i], false, false);
    }
    for (int i = 0; i < 4; ++i)
      for (int j = 0; j < 8; ++j)
        s_sc[wave][((half ? 8 : 0) + j) * 64 + i * 16 + l16] = acc[i][j];
    __syncthreads();
    if (lane < 16) {
      float mx = m_i;
      for (int j = 0; j < 64; ++j) mx = fmaxf(mx, s_sc[wave][lane * 64 + j]);
      float f = __expf(m_i - mx), sum = 0.f;
      for (int j = 0; j < 64; ++j) sum += __expf(s_sc[wave][lane * 64 + j] - mx);
      l_i = l_i * f + sum;
      m_i = mx;
    }
    __syncthreads();
  }
  if (lane < 16) {
    int row = r0 + lane;
    part_m[(size_t)row * VSPLIT + vs] = m_i;
    part_l[(size_t)row * VSPLIT + vs] = l_i;
  }
}

// =====================================================================
// 6) per-row loss: combine lse partials, bf16 dot for target logit,
//    weighted NLL into atomic accumulators
// =====================================================================
__global__ __launch_bounds__(64) void row_loss_kernel(
    const unsigned short* __restrict__ Hb, const unsigned short* __restrict__ Wt,
    const float* __restrict__ part_m, const float* __restrict__ part_l,
    const int* __restrict__ input_ids, const float* __restrict__ loss_mask,
    const int* __restrict__ anchors, const unsigned char* __restrict__ keep_mask,
    float* __restrict__ accum)
{
  int row = blockIdx.x;                       // 0..ROWS-1
  int b = row / QL, q = row % QL;
  int n = q / BSc, off = q % BSc;
  int anc = anchors[b * Nc + n];
  int lab = anc + off;
  bool keep = keep_mask[b * Nc + n] != 0;
  bool valid = lab < Sc;
  int safe = min(max(lab, 0), Sc - 1);
  float w = 0.f;
  if (keep && valid && off > 0)
    w = loss_mask[b * Sc + safe] * __expf(-(float)max(off - 1, 0) / GAMMAc);
  if (w == 0.f) return;                       // uniform across block -> safe

  int tgt = input_ids[b * Sc + safe];
  __shared__ float red[64];
  int t = threadIdx.x;
  const unsigned short* hr = Hb + (size_t)row * Dc;
  const unsigned short* wr = Wt + (size_t)tgt * Dc;
  float part = 0.f;
  for (int k = t; k < Dc; k += 64) part += bf2f(hr[k]) * bf2f(wr[k]);
  red[t] = part;
  __syncthreads();
  for (int s = 32; s > 0; s >>= 1) {
    if (t < s) red[t] += red[t + s];
    __syncthreads();
  }
  if (t == 0) {
    float M = -3.0e38f;
    for (int vsp = 0; vsp < VSPLIT; ++vsp)
      M = fmaxf(M, part_m[(size_t)row * VSPLIT + vsp]);
    float L = 0.f;
    for (int vsp = 0; vsp < VSPLIT; ++vsp)
      L += part_l[(size_t)row * VSPLIT + vsp] *
           __expf(part_m[(size_t)row * VSPLIT + vsp] - M);
    float lse = M + __logf(fmaxf(L, 1e-30f));
    float nll = lse - red[0];
    atomicAdd(&accum[0], w * nll);
    atomicAdd(&accum[1], w);
  }
}

__global__ void finalize_kernel(const float* accum, float* out) {
  out[0] = accum[0] / fmaxf(accum[1], 1e-6f);
}

// =====================================================================
// host launcher
// =====================================================================
extern "C" void kernel_launch(void* const* d_in, const int* in_sizes, int n_in,
                              void* d_out, int out_size, void* d_ws, size_t ws_size,
                              hipStream_t stream) {
  const int*   input_ids = (const int*)d_in[0];
  const float* loss_mask = (const float*)d_in[1];
  const int*   anchors   = (const int*)d_in[2];
  const unsigned char* keep_mask = (const unsigned char*)d_in[3];  // jax bool = 1B
  const float* embed = (const float*)d_in[4];
  const float* Wq = (const float*)d_in[5];
  const float* Wk = (const float*)d_in[6];
  const float* Wv = (const float*)d_in[7];
  const float* Wo = (const float*)d_in[8];
  const float* W_lm = (const float*)d_in[9];
  float* out = (float*)d_out;

  char* ws = (char*)d_ws;
  size_t off = 0;
  auto alloc = [&](size_t bytes) -> size_t {
    size_t o = off; off += (bytes + 255) & ~(size_t)255; return o;
  };
  unsigned short* kv   = (unsigned short*)(ws + alloc((size_t)Bc * KVL * Dc * 2));
  unsigned short* wtq  = (unsigned short*)(ws + alloc((size_t)Dc * Dc * 2));
  unsigned short* wtk  = (unsigned short*)(ws + alloc((size_t)Dc * Dc * 2));
  unsigned short* wtv  = (unsigned short*)(ws + alloc((size_t)Dc * Dc * 2));
  unsigned short* wto  = (unsigned short*)(ws + alloc((size_t)Dc * Dc * 2));
  unsigned short* wtlm = (unsigned short*)(ws + alloc((size_t)Vc * Dc * 2));
  unsigned short* qb   = (unsigned short*)(ws + alloc((size_t)Bc * QL * Dc * 2));
  unsigned short* kb   = (unsigned short*)(ws + alloc((size_t)Bc * KVL * Dc * 2));
  unsigned short* vT   = (unsigned short*)(ws + alloc((size_t)Bc * Dc * KVL * 2));
  unsigned short* attn = (unsigned short*)(ws + alloc((size_t)ROWS * Dc * 2));
  unsigned short* hid  = (unsigned short*)(ws + alloc((size_t)ROWS * Dc * 2));
  float* part_m = (float*)(ws + alloc((size_t)ROWS * VSPLIT * 4));
  float* part_l = (float*)(ws + alloc((size_t)ROWS * VSPLIT * 4));
  float* accum  = (float*)(ws + alloc(2 * 4));
  (void)ws_size; (void)in_sizes; (void)n_in; (void)out_size;

  zero_accum_kernel<<<1, 32, 0, stream>>>(accum);

  // 1) embeddings -> kv bf16
  prep_kv_kernel<<<Bc * KVL, 256, 0, stream>>>(input_ids, anchors, keep_mask,
                                               embed, kv);
  // 2) weight transpose+convert  (W[K][N] -> Wt[N][K] bf16)
  dim3 tb(32, 8);
  tconv_kernel<<<dim3(Dc / 32, Dc / 32), tb, 0, stream>>>(Wq, wtq, Dc, Dc);
  tconv_kernel<<<dim3(Dc / 32, Dc / 32), tb, 0, stream>>>(Wk, wtk, Dc, Dc);
  tconv_kernel<<<dim3(Dc / 32, Dc / 32), tb, 0, stream>>>(Wv, wtv, Dc, Dc);
  tconv_kernel<<<dim3(Dc / 32, Dc / 32), tb, 0, stream>>>(Wo, wto, Dc, Dc);
  tconv_kernel<<<dim3(Vc / 32, Dc / 32), tb, 0, stream>>>(W_lm, wtlm, Dc, Vc);

  // 3) projections (WMMA): q = noise_emb @ Wq ; k,v over full kv_in; v stored ^T
  gemm_bf16_kernel<<<dim3(Dc / 64, QL / 64, Bc), 32, 0, stream>>>(
      kv + (size_t)Sc * Dc, wtq, qb, QL, Dc, Dc,
      (long)KVL * Dc, (long)QL * Dc, 0, 0);
  gemm_bf16_kernel<<<dim3(Dc / 64, KVL / 64, Bc), 32, 0, stream>>>(
      kv, wtk, kb, KVL, Dc, Dc, (long)KVL * Dc, (long)KVL * Dc, 0, 0);
  gemm_bf16_kernel<<<dim3(Dc / 64, KVL / 64, Bc), 32, 0, stream>>>(
      kv, wtv, vT, KVL, Dc, Dc, (long)KVL * Dc, (long)Dc * KVL, 1, KVL);

  // 4) masked flash attention
  attn_kernel<<<dim3(Nc, Hc, Bc), 32, 0, stream>>>(qb, kb, vT, anchors, attn);

  // 5) output projection: hid = attn @ Wo
  gemm_bf16_kernel<<<dim3(Dc / 64, ROWS / 64, 1), 32, 0, stream>>>(
      attn, wto, hid, ROWS, Dc, Dc, 0, 0, 0, 0);

  // 6) LM head + streaming logsumexp partials (64 cols per pass)
  lm_lse_kernel<<<dim3(ROWS / 64, VSPLIT), 128, 0, stream>>>(hid, wtlm,
                                                             part_m, part_l);
  // 7) per-row loss
  row_loss_kernel<<<ROWS, 64, 0, stream>>>(hid, wtlm, part_m, part_l, input_ids,
                                           loss_mask, anchors, keep_mask, accum);
  // 8) finalize
  finalize_kernel<<<1, 1, 0, stream>>>(accum, out);
}
